// GraphAttentionPoolingWithGRU_82729660056044
// MI455X (gfx1250) — compile-verified, matrix operand order checked by reading, not run on hardware
//
#include <hip/hip_runtime.h>
#include <hip/hip_bf16.h>

// ---------------------------------------------------------------------------
// GraphAttentionPoolingWithGRU for MI455X (gfx1250, wave32, WMMA)
// B=64, N=512, H=512, HC=256, G3=768, G6=1536
//
// v3: batch all B-fragment loads of a k-step into one clause before the WMMA
// burst so waits stagger (loadcnt 7..0) and loads overlap matrix math.
// ---------------------------------------------------------------------------

typedef __attribute__((ext_vector_type(16))) __bf16         v16bf;
typedef __attribute__((ext_vector_type(8)))  float          v8f;
typedef __attribute__((ext_vector_type(8)))  unsigned short us8;

#define Bv 64
#define Nv 512
#define Hv 512
#define HCv 256
#define G3v 768
#define G6v 1536

__device__ __forceinline__ unsigned short f2bf(float f) {
  unsigned int u = __float_as_uint(f);
  unsigned int r = u + 0x7FFFu + ((u >> 16) & 1u);  // round-nearest-even
  return (unsigned short)(r >> 16);
}

// ---------------- 1) mask ----------------
__global__ __launch_bounds__(256) void mask_kernel(const float* __restrict__ x,
                                                   const int* __restrict__ nn,
                                                   float* __restrict__ xp) {
  long i = (long)blockIdx.x * 256 + threadIdx.x;   // B*N*H = 16M threads
  int b = (int)(i >> 18);                          // / (N*H)
  int t = (int)((i >> 9) & 511);                   // / H % N
  float v = x[i];
  xp[i] = (t < nn[b]) ? v : 0.0f;
}

// -------- 2) pack f32 matrix -> bf16 WMMA fragment order (A and B alike) ----
// src: [G rows, K cols] row-major.  dst[nt][kiter][lane][slot]:
//   n = nt*16 + lane%16, half = lane/16,
//   k = kiter*32 + (slot/8)*16 + half*8 + slot%8
__global__ __launch_bounds__(256) void pack_kernel(const float* __restrict__ src,
                                                   unsigned short* __restrict__ dst,
                                                   int G, int K) {
  int p = blockIdx.x * 256 + threadIdx.x;
  if (p >= G * K) return;
  int KIT = K >> 5;
  int per_nt = KIT * 512;
  int nt = p / per_nt;
  int r  = p - nt * per_nt;
  int kiter = r >> 9;
  int q = r & 511;
  int lane = q >> 4, slot = q & 15;
  int n = nt * 16 + (lane & 15);
  int half = lane >> 4;
  int k = kiter * 32 + ((slot >> 3) << 4) + half * 8 + (slot & 7);
  dst[p] = f2bf(src[(long)n * K + k]);
}

// ---------------- 3/5) gate GEMM: [32768,512] x [512,1536] ----------------
// One wave computes a 16x128 output strip: 8 N-tiles reuse one A fragment.
// Per k-step: clause of 18 b128 loads, then 8 WMMAs with staggered waits.
__global__ __launch_bounds__(256) void gemm_gates(const unsigned short* __restrict__ pA,
                                                  const unsigned short* __restrict__ pB,
                                                  const float* __restrict__ bias,
                                                  float* __restrict__ out) {
  const int KIT = 16;                               // 512 / 32
  const int NT  = 8;                                // N-tiles per wave
  int wid  = (blockIdx.x * 256 + threadIdx.x) >> 5; // global wave id
  int lane = threadIdx.x & 31;
  int mt = wid / 12;                                // 2048 M-tiles
  int ng = wid % 12;                                // 12 groups of 128 cols
  int mrow = lane & 15, half = lane >> 4;

  v8f acc[NT];
#pragma unroll
  for (int t = 0; t < NT; t++)
#pragma unroll
    for (int i = 0; i < 8; i++) acc[t][i] = 0.0f;

  for (int kit = 0; kit < KIT; kit++) {
    long aoff = (((long)mt * KIT + kit) * 32 + lane) * 16;
    v16bf av = *(const v16bf*)(pA + aoff);
    v16bf bv[NT];
#pragma unroll
    for (int t = 0; t < NT; t++) {
      long boff = ((((long)(ng * NT + t)) * KIT + kit) * 32 + lane) * 16;
      bv[t] = *(const v16bf*)(pB + boff);
    }
#pragma unroll
    for (int t = 0; t < NT; t++)
      acc[t] = __builtin_amdgcn_wmma_f32_16x16x32_bf16(
          false, av, false, bv[t], (short)0, acc[t], false, false);
  }
#pragma unroll
  for (int t = 0; t < NT; t++) {
    int g = ng * 128 + t * 16 + mrow;
    float bsc = bias[g];
#pragma unroll
    for (int v = 0; v < 8; v++) {
      long row = (long)mt * 16 + v + 8 * half;
      out[row * G6v + g] = acc[t][v] + bsc;
    }
  }
}

// ---------------- 4/6) recurrent scan (one dir, 16 batches / WG) ----------------
// grid = 8 blocks: blockIdx.x = dir*4 + batch_group. 256 threads (8 waves).
// h kept in LDS twice: f32 (for the z*h_old term) and bf16 fragment shadow.
__global__ __launch_bounds__(256) void scan_kernel(const float* __restrict__ xg,   // [B,N,1536]
                                                   const unsigned short* __restrict__ pW, // layer base
                                                   const float* __restrict__ bhh,  // [2][768]
                                                   float* __restrict__ hout) {     // [B,N,512]
  __shared__ float hcur[16][HCv];
  __shared__ unsigned short hbf[16][HCv];
  __shared__ float hg[16][G3v];
  const int KIT = 8;                                // 256 / 32
  int tid = threadIdx.x;
  int dir = blockIdx.x >> 2;
  int bg  = blockIdx.x & 3;
  const unsigned short* pWd = pW + (long)dir * (48L * KIT * 512);
  const float* bh = bhh + dir * G3v;
  int wave = tid >> 5, lane = tid & 31;
  int mrow = lane & 15, half = lane >> 4;

  for (int i = tid; i < 16 * HCv; i += 256) {
    ((float*)hcur)[i] = 0.0f;
    ((unsigned short*)hbf)[i] = 0;                  // bf16(0) == 0
  }
  __syncthreads();

  for (int s = 0; s < Nv; s++) {
    int t = dir ? (Nv - 1 - s) : s;

    // Phase A: 8 A-fragments from the bf16 shadow (2x ds_load_b128 each),
    // then 6 N-tiles/wave; per tile: clause of 16 b128 loads + 8 WMMAs.
    union { v16bf v; us8 h[2]; } af[KIT];
#pragma unroll
    for (int kit = 0; kit < KIT; kit++) {
      const unsigned short* hp = &hbf[mrow][kit * 32 + half * 8];
      af[kit].h[0] = *(const us8*)(hp);
      af[kit].h[1] = *(const us8*)(hp + 16);
    }
#pragma unroll 1
    for (int j = 0; j < 6; j++) {
      int nt = wave * 6 + j;
      v16bf bv[KIT];
#pragma unroll
      for (int kit = 0; kit < KIT; kit++)
        bv[kit] = *(const v16bf*)(pWd + (((long)nt * KIT + kit) * 32 + lane) * 16);
      v8f acc;
#pragma unroll
      for (int i = 0; i < 8; i++) acc[i] = 0.0f;
#pragma unroll
      for (int kit = 0; kit < KIT; kit++)
        acc = __builtin_amdgcn_wmma_f32_16x16x32_bf16(
            false, af[kit].v, false, bv[kit], (short)0, acc, false, false);
#pragma unroll
      for (int v = 0; v < 8; v++) hg[v + 8 * half][nt * 16 + mrow] = acc[v];
    }
    __syncthreads();

    // Phase B: GRU gate math for 16*256 (b,hc) pairs.
    for (int e = tid; e < 16 * HCv; e += 256) {
      int bb = e >> 8, hc = e & 255;
      long brow = (long)(bg * 16 + bb);
      long base = (brow * Nv + t) * G6v + dir * G3v;
      float xr = xg[base + hc];
      float xz = xg[base + HCv + hc];
      float xn = xg[base + 2 * HCv + hc];
      float hr = hg[bb][hc]            + bh[hc];
      float hz = hg[bb][HCv + hc]      + bh[HCv + hc];
      float hn = hg[bb][2 * HCv + hc]  + bh[2 * HCv + hc];
      float r = 1.0f / (1.0f + __expf(-(xr + hr)));
      float z = 1.0f / (1.0f + __expf(-(xz + hz)));
      float a = xn + r * hn;
      float ex = __expf(2.0f * a);                  // tanh(a) = (e^2a-1)/(e^2a+1)
      float n = (ex - 1.0f) / (ex + 1.0f);
      float hold = hcur[bb][hc];
      float h2 = (1.0f - z) * n + z * hold;
      hcur[bb][hc] = h2;
      hbf[bb][hc] = f2bf(h2);
      hout[(brow * Nv + t) * Hv + dir * HCv + hc] = h2;
    }
    __syncthreads();
  }
}

// ---------------- 7) residual + LayerNorm + score dot ----------------
__global__ __launch_bounds__(256) void ln_score_kernel(const float* __restrict__ h2,
                                                       const float* __restrict__ xp,
                                                       const float* __restrict__ gamma,
                                                       const float* __restrict__ beta,
                                                       const float* __restrict__ watt,
                                                       float* __restrict__ hln,
                                                       float* __restrict__ scores) {
  __shared__ float red[256];
  long row = blockIdx.x;                 // B*N rows
  int tid = threadIdx.x;
  const float* hr = h2 + row * Hv;
  const float* xr = xp + row * Hv;
  float y0 = hr[tid] + xr[tid];
  float y1 = hr[tid + 256] + xr[tid + 256];
  red[tid] = y0 + y1; __syncthreads();
  for (int s = 128; s > 0; s >>= 1) { if (tid < s) red[tid] += red[tid + s]; __syncthreads(); }
  float mu = red[0] * (1.0f / 512.0f); __syncthreads();
  float d0 = y0 - mu, d1 = y1 - mu;
  red[tid] = d0 * d0 + d1 * d1; __syncthreads();
  for (int s = 128; s > 0; s >>= 1) { if (tid < s) red[tid] += red[tid + s]; __syncthreads(); }
  float rs = rsqrtf(red[0] * (1.0f / 512.0f) + 1e-5f); __syncthreads();
  float o0 = d0 * rs * gamma[tid] + beta[tid];
  float o1 = d1 * rs * gamma[tid + 256] + beta[tid + 256];
  hln[row * Hv + tid] = o0;
  hln[row * Hv + tid + 256] = o1;
  red[tid] = o0 * watt[tid] + o1 * watt[tid + 256]; __syncthreads();
  for (int s = 128; s > 0; s >>= 1) { if (tid < s) red[tid] += red[tid + s]; __syncthreads(); }
  if (tid == 0) scores[row] = red[0];
}

// ---------------- 8) leaky_relu + softmax + weighted pooling ----------------
__global__ __launch_bounds__(512) void attend_kernel(const float* __restrict__ hln,
                                                     const float* __restrict__ scores,
                                                     const float* __restrict__ watt,
                                                     const float* __restrict__ q,
                                                     float* __restrict__ out) {
  __shared__ float warr[512];
  __shared__ float red[512];
  int b = blockIdx.x, tid = threadIdx.x;
  red[tid] = q[b * Hv + tid] * watt[Hv + tid]; __syncthreads();
  for (int s = 256; s > 0; s >>= 1) { if (tid < s) red[tid] += red[tid + s]; __syncthreads(); }
  float qd = red[0]; __syncthreads();
  float sc = scores[b * Nv + tid] + qd;
  sc = sc > 0.0f ? sc : 0.01f * sc;                       // leaky_relu
  red[tid] = sc; __syncthreads();
  for (int s = 256; s > 0; s >>= 1) { if (tid < s) red[tid] = fmaxf(red[tid], red[tid + s]); __syncthreads(); }
  float mx = red[0]; __syncthreads();
  float e = __expf(sc - mx);
  warr[tid] = e; red[tid] = e; __syncthreads();
  for (int s = 256; s > 0; s >>= 1) { if (tid < s) red[tid] += red[tid + s]; __syncthreads(); }
  float inv = 1.0f / red[0];
  __syncthreads();
  float acc = 0.0f;
  const float* hb = hln + (long)b * Nv * Hv;
  for (int t = 0; t < Nv; t++) acc += hb[(long)t * Hv + tid] * warr[t];
  out[b * Hv + tid] = acc * inv;
}

// ---------------------------------------------------------------------------
extern "C" void kernel_launch(void* const* d_in, const int* in_sizes, int n_in,
                              void* d_out, int out_size, void* d_ws, size_t ws_size,
                              hipStream_t stream) {
  const float* x    = (const float*)d_in[0];
  const int*   nn   = (const int*)d_in[1];
  const float* wih  = (const float*)d_in[2];   // [2,2,768,512]
  const float* whh  = (const float*)d_in[3];   // [2,2,768,256]
  const float* bih  = (const float*)d_in[4];   // [2,2,768]
  const float* bhh  = (const float*)d_in[5];   // [2,2,768]
  const float* lng  = (const float*)d_in[6];
  const float* lnb  = (const float*)d_in[7];
  const float* watt = (const float*)d_in[8];   // [1024,1]
  const float* q    = (const float*)d_in[9];   // [64,512]
  float* out = (float*)d_out;

  float* w = (float*)d_ws;
  float* xp     = w;                      // 16,777,216 f32
  float* xg     = w + 16777216L;          // 50,331,648 f32
  float* h1     = w + 67108864L;          // 16,777,216 f32
  float* h2     = w + 83886080L;          // 16,777,216 f32
  float* scores = w + 100663296L;         //     32,768 f32
  unsigned short* pack  = (unsigned short*)(w + 100696064L);
  unsigned short* pWih0 = pack;                // 1536*512 bf16
  unsigned short* pWih1 = pack + 786432L;      // 1536*512 bf16
  unsigned short* pWhh  = pack + 1572864L;     // 4 * 768*256 bf16
  unsigned short* pA    = pack + 2359296L;     // 32768*512 bf16 (activations)

  // 1) mask
  mask_kernel<<<65536, 256, 0, stream>>>(x, nn, xp);

  // 2) pack weights (bf16 WMMA fragment order)
  pack_kernel<<<(1536 * 512 + 255) / 256, 256, 0, stream>>>(wih,            pWih0, 1536, 512);
  pack_kernel<<<(1536 * 512 + 255) / 256, 256, 0, stream>>>(wih + 786432L,  pWih1, 1536, 512);
  for (int ld = 0; ld < 4; ld++)
    pack_kernel<<<(768 * 256 + 255) / 256, 256, 0, stream>>>(whh + ld * 196608L,
                                                             pWhh + ld * 196608L, 768, 256);

  // 3) layer-0: pack activations, gates GEMM, scan
  pack_kernel<<<65536, 256, 0, stream>>>(xp, pA, 32768, 512);
  gemm_gates<<<3072, 256, 0, stream>>>(pA, pWih0, bih, xg);
  scan_kernel<<<8, 256, 0, stream>>>(xg, pWhh, bhh, h1);

  // 5) layer-1: pack activations, gates GEMM, scan
  pack_kernel<<<65536, 256, 0, stream>>>(h1, pA, 32768, 512);
  gemm_gates<<<3072, 256, 0, stream>>>(pA, pWih1, bih + 1536, xg);
  scan_kernel<<<8, 256, 0, stream>>>(xg, pWhh + 2L * 196608L, bhh + 1536, h2);

  // 7) residual + LN + score, 8) softmax pooling (hln reuses h1)
  float* hln = h1;
  ln_score_kernel<<<32768, 256, 0, stream>>>(h2, xp, lng, lnb, watt, hln, scores);
  attend_kernel<<<64, 512, 0, stream>>>(hln, scores, watt, q, out);
}
